// SGC_26714696581628
// MI455X (gfx1250) — matrix operand order repaired and checked by reading, not compiled, not run on hardware
//
#include <hip/hip_runtime.h>

typedef __attribute__((ext_vector_type(16))) __bf16 v16bf;
typedef __attribute__((ext_vector_type(8)))  float  v8f;

// ---------------------------------------------------------------------------
// Degree accumulation: deg[dst[e]] += w[e]
// ---------------------------------------------------------------------------
__global__ void k_deg(const int* __restrict__ dst, const float* __restrict__ w,
                      float* __restrict__ deg, int E) {
  int e = blockIdx.x * blockDim.x + threadIdx.x;
  if (e < E) atomicAdd(&deg[dst[e]], w[e]);
}

// deg -> dinv in place (self-loop weight 1 folded in)
__global__ void k_dinv(float* __restrict__ deg, int N) {
  int i = blockIdx.x * blockDim.x + threadIdx.x;
  if (i < N) {
    float d = deg[i] + 1.0f;
    deg[i] = (d > 0.0f) ? rsqrtf(d) : 0.0f;
  }
}

// Self-loop term of one propagation hop: y = dinv^2 * x   (total = N*64)
__global__ void k_self_init(const float* __restrict__ x, const float* __restrict__ dinv,
                            float* __restrict__ y, int total) {
  int i = blockIdx.x * blockDim.x + threadIdx.x;
  if (i < total) {
    float dv = dinv[i >> 6];
    y[i] = dv * dv * x[i];
  }
}

// Edge scatter: y[dst][f] += dinv[src]*w*dinv[dst] * x[src][f]
__global__ void k_scatter(const int* __restrict__ src, const int* __restrict__ dst,
                          const float* __restrict__ w, const float* __restrict__ dinv,
                          const float* __restrict__ x, float* __restrict__ y, int E) {
  int tid = blockIdx.x * blockDim.x + threadIdx.x;   // E*64 = 76.8M < 2^31
  int e = tid >> 6;
  int f = tid & 63;
  if (e < E) {
    int s = src[e];
    int d = dst[e];
    float nm = dinv[s] * w[e] * dinv[d];
    atomicAdd(&y[d * 64 + f], nm * x[s * 64 + f]);
  }
}

// Zero-pad W (rows x 64) to padrows x 64
__global__ void k_pad_w(const float* __restrict__ W, float* __restrict__ Wp,
                        int rows, int total) {
  int i = blockIdx.x * blockDim.x + threadIdx.x;
  if (i < total) Wp[i] = (i < rows * 64) ? W[i] : 0.0f;
}

// ---------------------------------------------------------------------------
// C = relu?( A @ W^T + b )  via v_wmma_f32_16x16x32_bf16
//   A: nrows x 64 (f32, converted to bf16 in-register)
//   W: (ncolTiles*16) x 64 (f32, zero-padded rows)
//   One wave per 16x16 output tile; K=64 -> two WMMA issues per tile.
// ---------------------------------------------------------------------------
__global__ void __launch_bounds__(256)
k_gemm_wmma(const float* __restrict__ A, const float* __restrict__ W,
            const float* __restrict__ bias, float* __restrict__ C,
            int nrows, int outdim, int ncolTiles, int relu) {
  const int lane = threadIdx.x & 31;
  const int wid  = blockIdx.x * (blockDim.x >> 5) + (threadIdx.x >> 5);
  const int rowTiles = nrows >> 4;
  if (wid >= rowTiles * ncolTiles) return;   // wave-uniform: EXEC stays all-1s

  const int mt  = wid / ncolTiles;
  const int nt  = wid - mt * ncolTiles;
  const int hi  = lane >> 4;                 // half-wave select
  const int l15 = lane & 15;

  // A frag: lane holds row (mt*16+l15), K in [kb,kb+8) and [kb+16,kb+24), kb = 8*hi
  // B frag: lane holds W row (nt*16+l15), K in [kb,kb+8) and [kb+16,kb+24), kb = 8*hi
  const float* arow = A + (mt * 16 + l15) * 64 + hi * 8;
  const float* brow = W + (nt * 16 + l15) * 64 + hi * 8;

  v8f acc = {};
#pragma unroll
  for (int k0 = 0; k0 < 64; k0 += 32) {
    const float* ap = arow + k0;
    const float* bp = brow + k0;
    v16bf af, bfr;
#pragma unroll
    for (int i = 0; i < 8; ++i) {
      af[i]     = (__bf16)ap[i];
      af[8 + i] = (__bf16)ap[16 + i];
    }
#pragma unroll
    for (int i = 0; i < 8; ++i) { bfr[i] = (__bf16)bp[i]; bfr[8 + i] = (__bf16)bp[16 + i]; }
    acc = __builtin_amdgcn_wmma_f32_16x16x32_bf16(false, af, false, bfr,
                                                  (short)0, acc, false, false);
  }

  // C layout: lane, VGPR r -> row M = r + 8*hi, col N = l15
  const int ccol = nt * 16 + l15;
  if (ccol < outdim) {
    const float bv = bias[ccol];
    const int mrow = mt * 16 + hi * 8;
#pragma unroll
    for (int r = 0; r < 8; ++r) {
      float v = acc[r] + bv;
      if (relu) v = fmaxf(v, 0.0f);
      C[(mrow + r) * outdim + ccol] = v;
    }
  }
}

// ---------------------------------------------------------------------------
extern "C" void kernel_launch(void* const* d_in, const int* in_sizes, int n_in,
                              void* d_out, int out_size, void* d_ws, size_t ws_size,
                              hipStream_t stream) {
  const float* x  = (const float*)d_in[0];   // N x 64
  const int*   ei = (const int*)  d_in[1];   // 2 x E
  const float* ew = (const float*)d_in[2];   // E
  const float* W1 = (const float*)d_in[3];   // 64 x 64
  const float* b1 = (const float*)d_in[4];   // 64
  const float* W2 = (const float*)d_in[5];   // 40 x 64
  const float* b2 = (const float*)d_in[6];   // 40

  const int N    = in_sizes[0] / 64;
  const int E    = in_sizes[2];
  const int dout = in_sizes[5] / 64;              // 40
  const int outTiles = (dout + 15) / 16;          // 3
  const int padrows  = outTiles * 16;             // 48

  const int* src = ei;
  const int* dst = ei + E;

  // workspace: dinv(N) | bufA(N*64) | bufB(N*64) | w2pad(48*64)
  float* dinv = (float*)d_ws;
  float* bufA = dinv + N;
  float* bufB = bufA + (size_t)N * 64;
  float* w2p  = bufB + (size_t)N * 64;

  const int B = 256;
  const int gN   = (N + B - 1) / B;
  const int gE   = (E + B - 1) / B;
  const int totND = N * 64;
  const int gND  = (totND + B - 1) / B;
  const unsigned gEF = (unsigned)(((long long)E * 64 + B - 1) / B);

  // normalization
  hipMemsetAsync(dinv, 0, (size_t)N * sizeof(float), stream);
  k_deg<<<gE, B, 0, stream>>>(dst, ew, dinv, E);
  k_dinv<<<gN, B, 0, stream>>>(dinv, N);

  // ---- layer 1: two hops (x -> bufA -> bufB), then WMMA linear + ReLU -> bufA
  k_self_init<<<gND, B, 0, stream>>>(x, dinv, bufA, totND);
  k_scatter  <<<gEF, B, 0, stream>>>(src, dst, ew, dinv, x, bufA, E);
  k_self_init<<<gND, B, 0, stream>>>(bufA, dinv, bufB, totND);
  k_scatter  <<<gEF, B, 0, stream>>>(src, dst, ew, dinv, bufA, bufB, E);

  const int tiles1 = (N / 16) * 4;
  k_gemm_wmma<<<(tiles1 + 7) / 8, 256, 0, stream>>>(bufB, W1, b1, bufA, N, 64, 4, 1);

  // ---- layer 2: two hops (bufA -> bufB -> bufA), then WMMA linear -> d_out
  k_self_init<<<gND, B, 0, stream>>>(bufA, dinv, bufB, totND);
  k_scatter  <<<gEF, B, 0, stream>>>(src, dst, ew, dinv, bufA, bufB, E);
  k_self_init<<<gND, B, 0, stream>>>(bufB, dinv, bufA, totND);
  k_scatter  <<<gEF, B, 0, stream>>>(src, dst, ew, dinv, bufB, bufA, E);

  k_pad_w<<<(padrows * 64 + B - 1) / B, B, 0, stream>>>(W2, w2p, dout, padrows * 64);

  const int tiles2 = (N / 16) * outTiles;
  k_gemm_wmma<<<(tiles2 + 7) / 8, 256, 0, stream>>>(bufA, w2p, b2, (float*)d_out,
                                                    N, dout, outTiles, 0);
}
